// MultiheadAttention_21835613733480
// MI455X (gfx1250) — matrix operand order repaired and checked by reading, not compiled, not run on hardware
//
#include <hip/hip_runtime.h>

#define DMODEL 1024
#define NHEADS 16
#define DK     64
#define SEQ    2048
#define BATCH  2

typedef __attribute__((ext_vector_type(16))) __bf16 v16bf;
typedef __attribute__((ext_vector_type(8)))  float  v8f;

union BFrag {
  unsigned        u[8];
  unsigned short  h[16];
  v16bf           v;
};

// fp32 -> bf16 bits, round-to-nearest-even
__device__ __forceinline__ unsigned short f2bf(float f) {
  unsigned u = __builtin_bit_cast(unsigned, f);
  u += 0x7fffu + ((u >> 16) & 1u);
  return (unsigned short)(u >> 16);
}

// A-matrix (16x32 bf16) per-lane K offset for VGPR v, lane-half hf:
// VGPR0-3: K={0..7}(lanes0-15) / {8..15}(lanes16-31); VGPR4-7: +16
__device__ __forceinline__ int koffA(int v, int hf) {
  return ((v & 4) << 2) + hf * 8 + ((v & 3) << 1);
}
// B-matrix (32x16 bf16): same K order as A: VGPR0-3: K={0..7}/{8..15}; VGPR4-7: +16
__device__ __forceinline__ int koffB(int v, int hf) {
  return ((v & 4) << 2) + hf * 8 + ((v & 3) << 1);
}

// ---------------------------------------------------------------------------
// Phase 0: one-shot fp32 -> packed bf16 conversion (x and the 4 weights).
// Keeps all later GEMM inner loops conversion-free and halves their bytes.
// ---------------------------------------------------------------------------
__global__ void __launch_bounds__(256)
cvt_bf16_kernel(const float* __restrict__ src, unsigned* __restrict__ dst, int n2)
{
  const int i = blockIdx.x * 256 + threadIdx.x;
  if (i < n2) {
    const float2 f = ((const float2*)src)[i];
    dst[i] = (unsigned)f2bf(f.x) | ((unsigned)f2bf(f.y) << 16);
  }
}

// ---------------------------------------------------------------------------
// Phase 1: fused Q/K/V projection (y = x @ W^T) + RoPE on Q,K.
// One wave computes a 16(M=seq) x 64(N=feature) tile: A-frag reused across
// 4 WMMAs per k-step. Q scaled by 1/sqrt(dk).
// Q,K stored [b,h,s,d] bf16; V stored [b,h,d,s] bf16.
// ---------------------------------------------------------------------------
__global__ void __launch_bounds__(128)
qkv_rope_kernel(const unsigned short* __restrict__ xb,
                const int*   __restrict__ tpos,
                const unsigned short* __restrict__ wqb,
                const unsigned short* __restrict__ wkb,
                const unsigned short* __restrict__ wvb,
                unsigned short* __restrict__ qs,
                unsigned short* __restrict__ ks,
                unsigned short* __restrict__ vt)
{
  const int lane  = threadIdx.x & 31;
  const int wv_id = threadIdx.x >> 5;
  const int mtile = blockIdx.x * 4 + wv_id;     // 0..255  (B*S/16)
  const int ntile = blockIdx.y;                  // 0..47   (3*D/64)
  const int hf  = lane >> 4;
  const int l16 = lane & 15;

  const int r0  = mtile * 16;                    // global row = b*S + s
  const int eg0 = ntile * 64;                    // 0..3071
  const int sel = eg0 / DMODEL;                  // 0=Q 1=K 2=V
  const int e0  = eg0 % DMODEL;                  // 64-aligned -> one head
  const int head = e0 / DK;
  const unsigned short* w = (sel == 0) ? wqb : (sel == 1) ? wkb : wvb;

  const size_t arow = (size_t)(r0 + l16) * DMODEL;   // A: lane holds row m=l16
  size_t brow[4];
  #pragma unroll
  for (int t = 0; t < 4; ++t)
    brow[t] = (size_t)(e0 + t * 16 + l16) * DMODEL;  // B: lane holds col n=l16

  v8f acc[4] = {};
  for (int kk = 0; kk < DMODEL; kk += 32) {
    __builtin_prefetch(xb + arow + kk + 256, 0, 0);  // stream-ahead hint
    BFrag a;
    #pragma unroll
    for (int v = 0; v < 8; ++v)
      a.u[v] = *(const unsigned*)(xb + arow + kk + koffA(v, hf));
    #pragma unroll
    for (int t = 0; t < 4; ++t) {
      BFrag bm;
      #pragma unroll
      for (int v = 0; v < 8; ++v)
        bm.u[v] = *(const unsigned*)(w + brow[t] + kk + koffB(v, hf));
      acc[t] = __builtin_amdgcn_wmma_f32_16x16x32_bf16(false, a.v, false, bm.v,
                                                       (short)0, acc[t], false, false);
    }
  }

  // ---- epilogue: RoPE (Q,K) + scale (Q) + bf16 store -----------------------
  const int b  = r0 / SEQ;
  const int s0 = r0 % SEQ;

  float pos[8];
  #pragma unroll
  for (int j = 0; j < 8; ++j) pos[j] = (float)tpos[s0 + j + 8 * hf];

  const float scl = (sel == 0) ? 0.125f : 1.0f;  // fold 1/sqrt(64) into Q

  #pragma unroll
  for (int t = 0; t < 4; ++t) {
    const int dloc = t * 16 + l16;               // 0..63 within head
    float rv[8];
    #pragma unroll
    for (int j = 0; j < 8; ++j) rv[j] = acc[t][j];

    if (sel < 2) {
      // inv_freq = theta^(-2i/dk), i = dloc/2 ; ln(10000) = 9.21034...
      const float inv_freq =
          __expf(-9.210340371976184f * (float)(2 * (dloc >> 1)) / (float)DK);
      #pragma unroll
      for (int j = 0; j < 8; ++j) {
        float sn, cs;
        __sincosf(pos[j] * inv_freq, &sn, &cs);
        const float partner = __shfl_xor(rv[j], 1, 32);  // (2i,2i+1) adjacent lanes
        rv[j] = (dloc & 1) ? (rv[j] * cs + partner * sn)
                           : (rv[j] * cs - partner * sn);
      }
    }
    #pragma unroll
    for (int j = 0; j < 8; ++j) {
      const int srow = s0 + j + 8 * hf;
      const unsigned short bits = f2bf(rv[j] * scl);
      if (sel == 0)
        qs[(((size_t)b * NHEADS + head) * SEQ + srow) * DK + dloc] = bits;
      else if (sel == 1)
        ks[(((size_t)b * NHEADS + head) * SEQ + srow) * DK + dloc] = bits;
      else
        vt[(((size_t)b * NHEADS + head) * DK + dloc) * SEQ + srow] = bits;
    }
  }
}

// ---------------------------------------------------------------------------
// Phase 2: flash attention. One wave = one (b,h) x 16-query tile.
// 32-key steps: S = Q K^T (4 WMMA), online softmax (shfl over 16-lane halves),
// P through LDS (C-layout -> A-layout), O += P V (4 WMMA).
// ---------------------------------------------------------------------------
__global__ void __launch_bounds__(128)
flash_attn_kernel(const unsigned short* __restrict__ qs,
                  const unsigned short* __restrict__ ks,
                  const unsigned short* __restrict__ vt,
                  unsigned short* __restrict__ ao)
{
  __shared__ __align__(16) unsigned short pbuf[4][16 * 32];

  const int lane  = threadIdx.x & 31;
  const int wv_id = threadIdx.x >> 5;
  const int gw = blockIdx.x * 4 + wv_id;
  const int qt = gw & (SEQ / 16 - 1);            // 0..127
  const int bh = gw >> 7;                         // 0..31
  const int hf  = lane >> 4;
  const int l16 = lane & 15;
  const int q0  = qt * 16;

  // Q as two A-fragments (d = 0..31, 32..63), pre-scaled by 1/sqrt(dk)
  BFrag qa[2];
  #pragma unroll
  for (int t = 0; t < 2; ++t)
    #pragma unroll
    for (int v = 0; v < 8; ++v) {
      const int d = t * 32 + koffA(v, hf);
      qa[t].u[v] =
          *(const unsigned*)(qs + (((size_t)bh * SEQ + q0 + l16) * DK + d));
    }

  float ml[8], ll[8];
  v8f o0 = {}, o1 = {}, o2 = {}, o3 = {};
  #pragma unroll
  for (int j = 0; j < 8; ++j) { ml[j] = -1e30f; ll[j] = 0.0f; }

  for (int kt = 0; kt < SEQ / 32; ++kt) {
    const int key0 = kt * 32;

    // ---- scores: S[16x32] = Q(16x64) x K^T(64x32) ----
    v8f s0 = {}, s1 = {};
    #pragma unroll
    for (int t = 0; t < 2; ++t) {
      BFrag kb0, kb1;
      #pragma unroll
      for (int v = 0; v < 8; ++v) {
        const int d = t * 32 + koffB(v, hf);
        kb0.u[v] = *(const unsigned*)(ks + (((size_t)bh * SEQ + key0 + l16) * DK + d));
        kb1.u[v] = *(const unsigned*)(ks + (((size_t)bh * SEQ + key0 + 16 + l16) * DK + d));
      }
      s0 = __builtin_amdgcn_wmma_f32_16x16x32_bf16(false, qa[t].v, false, kb0.v,
                                                   (short)0, s0, false, false);
      s1 = __builtin_amdgcn_wmma_f32_16x16x32_bf16(false, qa[t].v, false, kb1.v,
                                                   (short)0, s1, false, false);
    }

    // ---- online softmax: row max / exp / row sum ----
    float mx[8];
    #pragma unroll
    for (int j = 0; j < 8; ++j) {
      mx[j] = fmaxf(s0[j], s1[j]);
      #pragma unroll
      for (int off = 1; off < 16; off <<= 1)
        mx[j] = fmaxf(mx[j], __shfl_xor(mx[j], off, 32));
    }
    float alpha[8], rs[8];
    #pragma unroll
    for (int j = 0; j < 8; ++j) {
      const float mn = fmaxf(ml[j], mx[j]);
      alpha[j] = __expf(ml[j] - mn);
      ml[j]    = mn;
      const float p0 = __expf(s0[j] - mn);
      const float p1 = __expf(s1[j] - mn);
      rs[j] = p0 + p1;
      const int row = j + 8 * hf;
      pbuf[wv_id][row * 32 + l16]      = f2bf(p0);
      pbuf[wv_id][row * 32 + 16 + l16] = f2bf(p1);
    }
    #pragma unroll
    for (int j = 0; j < 8; ++j) {
      #pragma unroll
      for (int off = 1; off < 16; off <<= 1)
        rs[j] += __shfl_xor(rs[j], off, 32);
      ll[j] = ll[j] * alpha[j] + rs[j];
      o0[j] *= alpha[j]; o1[j] *= alpha[j];
      o2[j] *= alpha[j]; o3[j] *= alpha[j];
    }

    // LDS C-layout stores must land before A-layout reads (cross-lane in-wave)
    asm volatile("s_wait_dscnt 0" ::: "memory");

    // ---- P as A-fragment (16x32 over keys) ----
    BFrag pa;
    #pragma unroll
    for (int v = 0; v < 8; ++v)
      pa.u[v] = *(const unsigned*)&pbuf[wv_id][l16 * 32 + koffA(v, hf)];

    // ---- O[16x64] += P x V (V^T stored [b,h,d,s] -> contiguous K-pairs) ----
    #pragma unroll
    for (int dt = 0; dt < 4; ++dt) {
      BFrag vb;
      #pragma unroll
      for (int v = 0; v < 8; ++v) {
        const int d = dt * 16 + l16;
        const int k = koffB(v, hf);
        vb.u[v] = *(const unsigned*)(vt + (((size_t)bh * DK + d) * SEQ + key0 + k));
      }
      v8f& oc = (dt == 0) ? o0 : (dt == 1) ? o1 : (dt == 2) ? o2 : o3;
      oc = __builtin_amdgcn_wmma_f32_16x16x32_bf16(false, pa.v, false, vb.v,
                                                   (short)0, oc, false, false);
    }
  }

  // ---- normalize + store attn output bf16 at [b, s, h*64 + d] ----
  const int b = bh >> 4, h = bh & 15;
  #pragma unroll
  for (int j = 0; j < 8; ++j) {
    const float inv  = 1.0f / ll[j];
    const int   srow = q0 + j + 8 * hf;
    unsigned short* dst = ao + ((size_t)b * SEQ + srow) * DMODEL + h * DK;
    dst[0 * 16 + l16] = f2bf(o0[j] * inv);
    dst[1 * 16 + l16] = f2bf(o1[j] * inv);
    dst[2 * 16 + l16] = f2bf(o2[j] * inv);
    dst[3 * 16 + l16] = f2bf(o3[j] * inv);
  }
}

// ---------------------------------------------------------------------------
// Phase 3: output projection: out = AO(bf16) @ Wo^T, fp32 result.
// One wave computes a 16 x 64 tile (4 accumulators).
// ---------------------------------------------------------------------------
__global__ void __launch_bounds__(128)
out_proj_kernel(const unsigned short* __restrict__ ao,
                const unsigned short* __restrict__ wob,
                float* __restrict__ out)
{
  const int lane  = threadIdx.x & 31;
  const int wv_id = threadIdx.x >> 5;
  const int mtile = blockIdx.x * 4 + wv_id;   // 0..255
  const int ntile = blockIdx.y;               // 0..15  (D/64)
  const int hf  = lane >> 4;
  const int l16 = lane & 15;

  const size_t arow = (size_t)(mtile * 16 + l16) * DMODEL;
  size_t brow[4];
  #pragma unroll
  for (int t = 0; t < 4; ++t)
    brow[t] = (size_t)(ntile * 64 + t * 16 + l16) * DMODEL;

  v8f acc[4] = {};
  for (int kk = 0; kk < DMODEL; kk += 32) {
    __builtin_prefetch(ao + arow + kk + 256, 0, 0);
    BFrag a;
    #pragma unroll
    for (int v = 0; v < 8; ++v)
      a.u[v] = *(const unsigned*)(ao + arow + kk + koffA(v, hf));
    #pragma unroll
    for (int t = 0; t < 4; ++t) {
      BFrag bm;
      #pragma unroll
      for (int v = 0; v < 8; ++v)
        bm.u[v] = *(const unsigned*)(wob + brow[t] + kk + koffB(v, hf));
      acc[t] = __builtin_amdgcn_wmma_f32_16x16x32_bf16(false, a.v, false, bm.v,
                                                       (short)0, acc[t], false, false);
    }
  }
  #pragma unroll
  for (int t = 0; t < 4; ++t)
    #pragma unroll
    for (int j = 0; j < 8; ++j) {
      const int row = mtile * 16 + j + 8 * hf;
      out[(size_t)row * DMODEL + ntile * 64 + t * 16 + l16] = acc[t][j];
    }
}

// ---------------------------------------------------------------------------
extern "C" void kernel_launch(void* const* d_in, const int* in_sizes, int n_in,
                              void* d_out, int out_size, void* d_ws, size_t ws_size,
                              hipStream_t stream) {
  (void)in_sizes; (void)n_in; (void)out_size; (void)ws_size;
  const float* x  = (const float*)d_in[0];
  const int*   tp = (const int*)d_in[1];
  const float* wq = (const float*)d_in[2];
  const float* wk = (const float*)d_in[3];
  const float* wv = (const float*)d_in[4];
  const float* wo = (const float*)d_in[5];
  float* out = (float*)d_out;

  const size_t nX = (size_t)BATCH * SEQ * DMODEL;            // 4M elems
  const size_t nW = (size_t)DMODEL * DMODEL;                  // 1M elems
  const size_t nPerMat = (size_t)BATCH * NHEADS * SEQ * DK;   // 4M elems

  unsigned short* xb  = (unsigned short*)d_ws;
  unsigned short* wqb = xb  + nX;
  unsigned short* wkb = wqb + nW;
  unsigned short* wvb = wkb + nW;
  unsigned short* wob = wvb + nW;
  unsigned short* qs  = wob + nW;
  unsigned short* ks  = qs  + nPerMat;
  unsigned short* vt  = ks  + nPerMat;
  unsigned short* ao  = vt  + nPerMat;   // total ~48 MB workspace (bf16)

  // Phase 0: bf16 conversions
  cvt_bf16_kernel<<<(int)(nX / 2 / 256), 256, 0, stream>>>(x,  (unsigned*)xb,  (int)(nX / 2));
  cvt_bf16_kernel<<<(int)(nW / 2 / 256), 256, 0, stream>>>(wq, (unsigned*)wqb, (int)(nW / 2));
  cvt_bf16_kernel<<<(int)(nW / 2 / 256), 256, 0, stream>>>(wk, (unsigned*)wkb, (int)(nW / 2));
  cvt_bf16_kernel<<<(int)(nW / 2 / 256), 256, 0, stream>>>(wv, (unsigned*)wvb, (int)(nW / 2));
  cvt_bf16_kernel<<<(int)(nW / 2 / 256), 256, 0, stream>>>(wo, (unsigned*)wob, (int)(nW / 2));

  dim3 blk(128);

  dim3 g1((BATCH * SEQ / 16) / 4, 3 * DMODEL / 64);       // (64, 48)
  qkv_rope_kernel<<<g1, blk, 0, stream>>>(xb, tp, wqb, wkb, wvb, qs, ks, vt);

  dim3 g2((BATCH * NHEADS * (SEQ / 16)) / 4);             // 1024
  flash_attn_kernel<<<g2, blk, 0, stream>>>(qs, ks, vt, ao);

  dim3 g3((BATCH * SEQ / 16) / 4, DMODEL / 64);           // (64, 16)
  out_proj_kernel<<<g3, blk, 0, stream>>>(ao, wob, out);
}
